// NLM_18116172055210
// MI455X (gfx1250) — matrix operand order repaired and checked by reading, not compiled, not run on hardware
//
#include <hip/hip_runtime.h>

typedef __attribute__((ext_vector_type(16))) _Float16 v16h;
typedef __attribute__((ext_vector_type(8)))  _Float16 v8h;
typedef __attribute__((ext_vector_type(8)))  float    v8f;

#define C_DIM   128     // channels (K of the GEMM)
#define N_DIM   4096    // 64*64 spatial positions
#define BATCH   4
#define QT      64      // columns per workgroup
#define PT      64      // rows per iteration
#define NT      128     // threads per block (4 waves of 32)
#define LSTR    136     // LDS row stride in halves (128 + 8 pad)
#define KC      (C_DIM / 32)              // K chunks per tile = 4
#define CHUNKS  ((PT * C_DIM / 8) / NT)   // v8h chunks per thread per 64x128 tile = 8

__device__ __forceinline__ v16h cat16(v8h a, v8h b) {
  return __builtin_shufflevector(a, b, 0,1,2,3,4,5,6,7,8,9,10,11,12,13,14,15);
}

// One-shot convert + transpose: f32 [b][c][p]  ->  f16 [b][p][c] (K-contiguous)
// blockIdx.z selects in1->o1 or in2->o2.
__global__ __launch_bounds__(NT) void cvt_kernel(const float* __restrict__ in1,
                                                 const float* __restrict__ in2,
                                                 _Float16* __restrict__ o1,
                                                 _Float16* __restrict__ o2) {
  __shared__ __align__(16) _Float16 tile[PT * LSTR];
  const int t  = threadIdx.x;
  const int b  = blockIdx.y;
  const int p0 = blockIdx.x * PT;
  const float*  src = (blockIdx.z ? in2 : in1) + (size_t)b * C_DIM * N_DIM;
  _Float16*     dst = (blockIdx.z ? o2  : o1 ) + (size_t)b * N_DIM * C_DIM;

  for (int idx4 = t; idx4 < (PT * C_DIM) / 4; idx4 += NT) {
    int c  = idx4 >> 4;
    int pl = (idx4 & 15) << 2;
    float4 v = *(const float4*)(src + (size_t)c * N_DIM + p0 + pl);
    tile[(pl + 0) * LSTR + c] = (_Float16)v.x;
    tile[(pl + 1) * LSTR + c] = (_Float16)v.y;
    tile[(pl + 2) * LSTR + c] = (_Float16)v.z;
    tile[(pl + 3) * LSTR + c] = (_Float16)v.w;
  }
  __syncthreads();
  #pragma unroll
  for (int i = 0; i < CHUNKS; ++i) {
    int idx = t + i * NT;
    int row = idx >> 4;
    int cc  = (idx & 15) << 3;
    *(v8h*)(dst + (size_t)(p0 + row) * C_DIM + cc) = *(const v8h*)(&tile[row * LSTR + cc]);
  }
}

// diag[b,p] = sum_c f1[b,c,p] * f2[b,c,p]   (fp32, coalesced across p)
__global__ __launch_bounds__(256) void diag_kernel(const float* __restrict__ f1,
                                                   const float* __restrict__ f2,
                                                   float* __restrict__ d) {
  int i = blockIdx.x * blockDim.x + threadIdx.x;      // i = b*N + p
  if (i >= BATCH * N_DIM) return;
  int b = i >> 12;
  int p = i & (N_DIM - 1);
  const float* a  = f1 + (size_t)b * C_DIM * N_DIM + p;
  const float* bb = f2 + (size_t)b * C_DIM * N_DIM + p;
  float acc = 0.f;
  #pragma unroll 8
  for (int c = 0; c < C_DIM; ++c)
    acc += a[(size_t)c * N_DIM] * bb[(size_t)c * N_DIM];
  d[i] = acc;
}

// PASS 1: stream all rows, online softmax stats per column (max, rescaled sum).
// PASS 2: recompute GEMM, write exp(v - colmax) / colsum.
// f1h/f2h are f16 [b][pos][c] (K-contiguous) produced by cvt_kernel.
template <int PASS>
__global__ __launch_bounds__(NT) void aff_kernel(const _Float16* __restrict__ f1h,
                                                 const _Float16* __restrict__ f2h,
                                                 const float* __restrict__ diag,
                                                 float* __restrict__ colmax,
                                                 float* __restrict__ colsum,
                                                 float* __restrict__ out) {
  __shared__ __align__(16) _Float16 sA[PT * LSTR];   // rows p_local x K
  __shared__ __align__(16) _Float16 sB[QT * LSTR];   // cols q_local x K
  __shared__ float sD[PT];
  __shared__ float sRedM[NT];
  __shared__ float sRedS[NT];

  const int t    = threadIdx.x;
  const int wave = t >> 5;
  const int lane = t & 31;
  const int hi   = lane >> 4;    // which half of the wave
  const int lc   = lane & 15;    // lane within half: output column / frag row
  const int b    = blockIdx.y;
  const int q0   = blockIdx.x * QT;

  const _Float16* F1 = f1h + (size_t)b * N_DIM * C_DIM;
  const _Float16* F2 = f2h + (size_t)b * N_DIM * C_DIM;

  // Stage B tile once: QT columns x C_DIM K (b128 load + b128 LDS store)
  #pragma unroll
  for (int i = 0; i < CHUNKS; ++i) {
    int idx = t + i * NT;
    int row = idx >> 4;
    int cc  = (idx & 15) << 3;
    *(v8h*)(&sB[row * LSTR + cc]) =
        *(const v8h*)(F2 + (size_t)(q0 + row) * C_DIM + cc);
  }

  const int q = q0 + wave * 16 + lc;     // this lane's output column
  float m = -3.0e38f, s = 0.f;           // pass-1 running stats
  float cm = 0.f, cinv = 0.f;            // pass-2 column stats
  if (PASS == 2) {
    cm   = colmax[b * N_DIM + q];
    cinv = 1.0f / colsum[b * N_DIM + q];
  }

  // Per-lane fragment base pointers.
  // B frag (32x16): lane lc = column, halves K = kc*32 + hi*16 + [0..15]
  // A frag (16x32): lane lc = row,    halves K = kc*32 + hi*16 + [0..15]
  const _Float16* sBp = &sB[(wave * 16 + lc) * LSTR + hi * 16];
  const _Float16* sAp = &sA[lc * LSTR + hi * 16];
#define LOADB(kc) cat16(*(const v8h*)(sBp + (kc) * 32), \
                        *(const v8h*)(sBp + (kc) * 32 + 8))
#define LOADA(rt, kc) cat16(*(const v8h*)(sAp + (rt) * 16 * LSTR + (kc) * 32), \
                            *(const v8h*)(sAp + (rt) * 16 * LSTR + (kc) * 32 + 8))

  // Software pipeline: A tile for the *next* iteration lives in registers.
  v8h  pre[CHUNKS];
  float dpre = (t < PT) ? diag[b * N_DIM + t] : 0.f;
  #pragma unroll
  for (int i = 0; i < CHUNKS; ++i) {
    int idx = t + i * NT;
    pre[i] = *(const v8h*)(F1 + (size_t)(idx >> 4) * C_DIM + ((idx & 15) << 3));
  }

  for (int p0 = 0; p0 < N_DIM; p0 += PT) {
    __syncthreads();   // previous iteration's readers of sA/sD are done
    #pragma unroll
    for (int i = 0; i < CHUNKS; ++i) {
      int idx = t + i * NT;
      *(v8h*)(&sA[(idx >> 4) * LSTR + ((idx & 15) << 3)]) = pre[i];
    }
    if (t < PT) sD[t] = dpre;
    if (p0 + PT < N_DIM) {           // issue next tile's loads; wait happens next iter
      #pragma unroll
      for (int i = 0; i < CHUNKS; ++i) {
        int idx = t + i * NT;
        pre[i] = *(const v8h*)(F1 + (size_t)(p0 + PT + (idx >> 4)) * C_DIM +
                               ((idx & 15) << 3));
      }
      dpre = (t < PT) ? diag[b * N_DIM + p0 + PT + t] : 0.f;
      if (p0 + 2 * PT < N_DIM)       // warm L2 for tile i+2 (global_prefetch_b8)
        __builtin_prefetch((const char*)F1 + (size_t)(p0 + 2 * PT) * C_DIM * 2 +
                           t * 128, 0, 0);
    }
    __syncthreads();

    // K loop with register-double-buffered fragments: loads for chunk kc+1 are
    // issued BEFORE the 4 WMMAs of chunk kc, so the WMMA burst hides ds latency.
    v8f acc[4] = {};   // 4 row tiles of 16 -> 64 rows x 16 cols per wave
    v16h bf  = LOADB(0);
    v16h af0 = LOADA(0, 0), af1 = LOADA(1, 0), af2 = LOADA(2, 0), af3 = LOADA(3, 0);
    #pragma unroll
    for (int kc = 0; kc < KC; ++kc) {
      v16h bfn = bf, an0 = af0, an1 = af1, an2 = af2, an3 = af3;
      if (kc + 1 < KC) {
        bfn = LOADB(kc + 1);
        an0 = LOADA(0, kc + 1);
        an1 = LOADA(1, kc + 1);
        an2 = LOADA(2, kc + 1);
        an3 = LOADA(3, kc + 1);
      }
      acc[0] = __builtin_amdgcn_wmma_f32_16x16x32_f16(false, af0, false, bf,
                                                      (short)0, acc[0], false, false);
      acc[1] = __builtin_amdgcn_wmma_f32_16x16x32_f16(false, af1, false, bf,
                                                      (short)0, acc[1], false, false);
      acc[2] = __builtin_amdgcn_wmma_f32_16x16x32_f16(false, af2, false, bf,
                                                      (short)0, acc[2], false, false);
      acc[3] = __builtin_amdgcn_wmma_f32_16x16x32_f16(false, af3, false, bf,
                                                      (short)0, acc[3], false, false);
      bf = bfn; af0 = an0; af1 = an1; af2 = an2; af3 = an3;
    }

    // C/D layout: acc[rt][j] -> row p0 + rt*16 + hi*8 + j, col q
    if (PASS == 1) {
      float tmax = -3.0e38f;
      #pragma unroll
      for (int rt = 0; rt < 4; ++rt)
        #pragma unroll
        for (int j = 0; j < 8; ++j) {
          float v = acc[rt][j] - 0.5f * sD[rt * 16 + hi * 8 + j];
          tmax = fmaxf(tmax, v);
        }
      float nm = fmaxf(m, tmax);
      float ls = 0.f;
      #pragma unroll
      for (int rt = 0; rt < 4; ++rt)
        #pragma unroll
        for (int j = 0; j < 8; ++j) {
          float v = acc[rt][j] - 0.5f * sD[rt * 16 + hi * 8 + j];
          ls += __expf(v - nm);
        }
      s = s * __expf(m - nm) + ls;
      m = nm;
    } else {
      float* ocol = out + (size_t)b * N_DIM * N_DIM + (size_t)p0 * N_DIM + q;
      #pragma unroll
      for (int rt = 0; rt < 4; ++rt)
        #pragma unroll
        for (int j = 0; j < 8; ++j) {
          float v = acc[rt][j] - 0.5f * sD[rt * 16 + hi * 8 + j];
          ocol[(size_t)(rt * 16 + hi * 8 + j) * N_DIM] = __expf(v - cm) * cinv;
        }
    }
  }
#undef LOADA
#undef LOADB

  if (PASS == 1) {
    // Combine the two half-wave partials (same column, disjoint rows)
    sRedM[t] = m; sRedS[t] = s;
    __syncthreads();
    if (hi == 0) {
      float m2 = sRedM[t + 16], s2 = sRedS[t + 16];
      float nm = fmaxf(m, m2);
      float ss = s * __expf(m - nm) + s2 * __expf(m2 - nm);
      colmax[b * N_DIM + q] = nm;
      colsum[b * N_DIM + q] = ss;
    }
  }
}

extern "C" void kernel_launch(void* const* d_in, const int* in_sizes, int n_in,
                              void* d_out, int out_size, void* d_ws, size_t ws_size,
                              hipStream_t stream) {
  const float* in1 = (const float*)d_in[0];
  const float* in2 = (const float*)d_in[1];
  float* out = (float*)d_out;

  // Workspace layout:
  //   diag   : BATCH*N f32
  //   colmax : BATCH*N f32
  //   colsum : BATCH*N f32
  //   f1h    : BATCH*N*C f16
  //   f2h    : BATCH*N*C f16
  float* dws  = (float*)d_ws;
  float* cmax = dws  + BATCH * N_DIM;
  float* csum = cmax + BATCH * N_DIM;
  _Float16* f1h = (_Float16*)(csum + BATCH * N_DIM);
  _Float16* f2h = f1h + (size_t)BATCH * N_DIM * C_DIM;

  dim3 cgrid(N_DIM / PT, BATCH, 2);
  cvt_kernel<<<cgrid, NT, 0, stream>>>(in1, in2, f1h, f2h);

  diag_kernel<<<(BATCH * N_DIM) / 256, 256, 0, stream>>>(in1, in2, dws);

  dim3 grid(N_DIM / QT, BATCH);
  aff_kernel<1><<<grid, NT, 0, stream>>>(f1h, f2h, dws, cmax, csum, out);
  aff_kernel<2><<<grid, NT, 0, stream>>>(f1h, f2h, dws, cmax, csum, out);
}
